// FusedAttentionBlock_1975684956171
// MI455X (gfx1250) — matrix operand order repaired and checked by reading, not compiled, not run on hardware
//
#include <hip/hip_runtime.h>
#include <stdint.h>

// ---------------------------------------------------------------------------
// FusedAttentionBlock for MI455X (gfx1250, wave32, WMMA, async-to-LDS)
//   x:[2,2048,1024] f32, w_qkv:[3072,1024] f32, w_dense:[1024,1024] f32
//   out:[2,2048,1024] f32
// Strategy: convert once to bf16, run all matmuls through
// v_wmma_f32_16x16x32_bf16 with f32 accumulate; stage tiles with
// global_load_async_to_lds_b128 (ASYNCcnt) and double-buffer in the GEMM.
// ---------------------------------------------------------------------------

typedef __bf16 bf16x16 __attribute__((ext_vector_type(16)));
typedef float  floatx8 __attribute__((ext_vector_type(8)));

union Frag16 {               // one WMMA 16-bit operand: 8 VGPRs
    uint4   u[2];
    bf16x16 v;
};

static __device__ inline floatx8 fzero8() {
    floatx8 v;
#pragma unroll
    for (int i = 0; i < 8; ++i) v[i] = 0.0f;
    return v;
}

static __device__ inline uint16_t f2bf(float f) {
    union { float f; uint32_t u; } c; c.f = f;
    uint32_t u = c.u;
    uint32_t r = u + 0x7FFFu + ((u >> 16) & 1u);   // round-to-nearest-even
    return (uint16_t)(r >> 16);
}

static __device__ inline floatx8 wmma_bf16(bf16x16 a, bf16x16 b, floatx8 c) {
    return __builtin_amdgcn_wmma_f32_16x16x32_bf16(
        false, a, false, b, (short)0, c, false, false);
}

// generic LDS pointer -> 32-bit LDS byte offset (low 32 bits of the
// shared-aperture generic address, per ISA flat->LDS truncation rule)
static __device__ inline uint32_t lds_off(const void* p) {
    return (uint32_t)(uintptr_t)p;
}

// one 16-byte async copy: LDS[ldsAddr] = *(saddr + voffByte)   (ASYNCcnt)
static __device__ inline void async_b128(uint32_t ldsAddr, uint32_t voffByte,
                                         const void* saddr) {
    asm volatile("global_load_async_to_lds_b128 %0, %1, %2"
                 :: "v"(ldsAddr), "v"(voffByte), "s"(saddr)
                 : "memory");
}

static __device__ inline void wait_async0() {
    asm volatile("s_wait_asynccnt 0x0" ::: "memory");
}

// ---------------------------------------------------------------------------
// f32 -> bf16 conversion (grid-stride)
// ---------------------------------------------------------------------------
__global__ __launch_bounds__(256) void cvt_f32_bf16(
    const float* __restrict__ in, uint16_t* __restrict__ out, int n) {
    int i = blockIdx.x * blockDim.x + threadIdx.x;
    int stride = gridDim.x * blockDim.x;
    for (; i < n; i += stride) out[i] = f2bf(in[i]);
}

// ---------------------------------------------------------------------------
// NT GEMM:  out[M,N] = A[M,K] * W[N,K]^T   (A,W bf16; acc f32)
// Block: 256 threads (8 waves), tile 128x128, K-step 64, double-buffered
// async-to-LDS staging. Wave (wm 0..3, wn 0..1) computes 2x4 WMMA tiles.
// ---------------------------------------------------------------------------
template <bool OUT_F32>
__global__ __launch_bounds__(256) void gemm_bf16_nt(
    const uint16_t* __restrict__ A, const uint16_t* __restrict__ W,
    void* __restrict__ outp, int M, int N, int K) {
    __shared__ __align__(16) uint16_t As[2][128 * 64];
    __shared__ __align__(16) uint16_t Bs[2][128 * 64];

    const int tid  = threadIdx.x;
    const int lane = tid & 31, wid = tid >> 5;
    const int wm   = wid & 3,  wn  = wid >> 2;
    const int lm   = lane & 15;
    const int hs   = lane >> 4;                 // half-wave select
    const int bn   = blockIdx.x, bm = blockIdx.y;

    const uint16_t* Abase = A + (size_t)bm * 128 * K;
    const uint16_t* Bbase = W + (size_t)bn * 128 * K;

    floatx8 acc[2][4];
#pragma unroll
    for (int mi = 0; mi < 2; ++mi)
#pragma unroll
        for (int ni = 0; ni < 4; ++ni) acc[mi][ni] = fzero8();

    // stage one 128x64 tile of A and W into buffer `buf` (async, 16B/lane x4)
    auto issue_tile = [&](int k0, int buf) {
#pragma unroll
        for (int i = 0; i < 4; ++i) {
            const int c   = tid * 4 + i;        // 1024 chunks of 8 bf16
            const int row = c >> 3, cc = c & 7;
            const uint32_t gb = (uint32_t)(((row * K) + k0 + cc * 8) * 2);
            async_b128(lds_off(&As[buf][row * 64 + cc * 8]), gb, Abase);
            async_b128(lds_off(&Bs[buf][row * 64 + cc * 8]), gb, Bbase);
        }
    };

    issue_tile(0, 0);
    int it = 0;
    for (int k0 = 0; k0 < K; k0 += 64, ++it) {
        const int buf = it & 1;
        wait_async0();                 // this wave's copies into `buf` done
        __syncthreads();               // everyone's copies done, prev reads done
        if (k0 + 64 < K) issue_tile(k0 + 64, buf ^ 1);   // overlap next copy

#pragma unroll
        for (int kk = 0; kk < 2; ++kk) {       // two 32-wide K steps
            Frag16 af[2];
#pragma unroll
            for (int mi = 0; mi < 2; ++mi) {
                const uint16_t* p =
                    &As[buf][(wm * 32 + mi * 16 + lm) * 64 + kk * 32 + hs * 8];
                af[mi].u[0] = *(const uint4*)p;
                af[mi].u[1] = *(const uint4*)(p + 16);
            }
            Frag16 bfm[4];
#pragma unroll
            for (int ni = 0; ni < 4; ++ni) {
                const uint16_t* p =
                    &Bs[buf][(wn * 64 + ni * 16 + lm) * 64 + kk * 32 + hs * 8];
                bfm[ni].u[0] = *(const uint4*)p;
                bfm[ni].u[1] = *(const uint4*)(p + 16);
            }
#pragma unroll
            for (int mi = 0; mi < 2; ++mi)
#pragma unroll
                for (int ni = 0; ni < 4; ++ni)
                    acc[mi][ni] = wmma_bf16(af[mi].v, bfm[ni].v, acc[mi][ni]);
        }
        __syncthreads();               // done reading `buf`
    }

    // C layout: VGPR r -> row r + hs*8, col = lm
#pragma unroll
    for (int mi = 0; mi < 2; ++mi) {
        const int rowb = bm * 128 + wm * 32 + mi * 16 + hs * 8;
#pragma unroll
        for (int ni = 0; ni < 4; ++ni) {
            const int col = bn * 128 + wn * 64 + ni * 16 + lm;
#pragma unroll
            for (int r = 0; r < 8; ++r) {
                if (OUT_F32)
                    ((float*)outp)[(size_t)(rowb + r) * N + col] = acc[mi][ni][r];
                else
                    ((uint16_t*)outp)[(size_t)(rowb + r) * N + col] = f2bf(acc[mi][ni][r]);
            }
        }
    }
}

// ---------------------------------------------------------------------------
// Flash attention: qkv [4096,3072] bf16 (cols: Q | K | V, per-head 64 wide)
// -> ctx [4096,1024] bf16. Grid: B*H*(S/128)=512 blocks, 8 waves/block,
// each wave owns 16 query rows; KV processed in chunks of 32.
// K tile staged with async-to-LDS; V staged transposed via registers.
// ---------------------------------------------------------------------------
__global__ __launch_bounds__(256) void attn_fwd(
    const uint16_t* __restrict__ qkv, uint16_t* __restrict__ ctx) {
    __shared__ __align__(16) uint16_t Ks[32 * 64];    // K chunk, row-major [kv][hd]
    __shared__ __align__(16) uint16_t Vts[64 * 32];   // V chunk transposed [hd][kv]
    __shared__ __align__(16) uint16_t Ps[8 * 16 * 32];// per-wave P scratch

    const int tid  = threadIdx.x;
    const int lane = tid & 31, wid = tid >> 5;
    const int lm   = lane & 15, hs = lane >> 4;
    const int bid  = blockIdx.x;
    const int b  = bid >> 8;
    const int h  = (bid >> 4) & 15;
    const int qb = bid & 15;
    const int qrowg = b * 2048 + qb * 128 + wid * 16;  // wave's first query row

    // Q fragments (16x64 over two K-steps of 32), straight from global
    Frag16 qf[2];
    {
        const uint16_t* qbase = qkv + (size_t)(qrowg + lm) * 3072 + h * 64;
#pragma unroll
        for (int kk = 0; kk < 2; ++kk) {
            const uint16_t* p = qbase + kk * 32 + hs * 8;
            qf[kk].u[0] = *(const uint4*)p;
            qf[kk].u[1] = *(const uint4*)(p + 16);
        }
    }

    float m_run[8], l_run[8];
#pragma unroll
    for (int r = 0; r < 8; ++r) { m_run[r] = -__builtin_inff(); l_run[r] = 0.0f; }
    floatx8 accv[4];
#pragma unroll
    for (int ni = 0; ni < 4; ++ni) accv[ni] = fzero8();

    const int krow = tid >> 3, kch = tid & 7;   // staging coords (32 rows x 8 chunks)
    const uint16_t* kv_base = qkv + (size_t)(b * 2048) * 3072 + h * 64;
    const uint32_t  koffLds = lds_off(&Ks[krow * 64 + kch * 8]);

    for (int kv0 = 0; kv0 < 2048; kv0 += 32) {
        const size_t srow = (size_t)(b * 2048 + kv0 + krow) * 3072 + h * 64 + kch * 8;
        uint4 kv_v = *(const uint4*)(qkv + srow + 2048);   // V via registers
        __syncthreads();                                   // prev tile consumed
        // K via async DMA to LDS
        async_b128(koffLds,
                   (uint32_t)((((kv0 + krow) * 3072) + 1024 + kch * 8) * 2),
                   kv_base);
        {   // transpose V into [hd][kv]
            union { uint4 u; uint16_t e[8]; } t; t.u = kv_v;
#pragma unroll
            for (int i = 0; i < 8; ++i) Vts[(kch * 8 + i) * 32 + krow] = t.e[i];
        }
        wait_async0();
        __syncthreads();

        // S = Q*K^T for two 16-wide kv tiles, accumulating over hd (2 steps)
        floatx8 sacc[2];
#pragma unroll
        for (int t = 0; t < 2; ++t) {
            floatx8 st = fzero8();
#pragma unroll
            for (int kk = 0; kk < 2; ++kk) {
                Frag16 kb;
                const uint16_t* p = &Ks[(t * 16 + lm) * 64 + kk * 32 + hs * 8];
                kb.u[0] = *(const uint4*)p;
                kb.u[1] = *(const uint4*)(p + 16);
                st = wmma_bf16(qf[kk].v, kb.v, st);
            }
            sacc[t] = st;
        }

        // online softmax: row reductions live inside each 16-lane half
#pragma unroll
        for (int r = 0; r < 8; ++r) {
            float v0 = sacc[0][r] * 0.125f;
            float v1 = sacc[1][r] * 0.125f;
            float mx = fmaxf(v0, v1);
            mx = fmaxf(mx, __shfl_xor(mx, 1, 32));
            mx = fmaxf(mx, __shfl_xor(mx, 2, 32));
            mx = fmaxf(mx, __shfl_xor(mx, 4, 32));
            mx = fmaxf(mx, __shfl_xor(mx, 8, 32));
            float mN    = fmaxf(m_run[r], mx);
            float alpha = __expf(m_run[r] - mN);
            float p0 = __expf(v0 - mN), p1 = __expf(v1 - mN);
            float sm = p0 + p1;
            sm += __shfl_xor(sm, 1, 32);
            sm += __shfl_xor(sm, 2, 32);
            sm += __shfl_xor(sm, 4, 32);
            sm += __shfl_xor(sm, 8, 32);
            l_run[r] = l_run[r] * alpha + sm;
            m_run[r] = mN;
            accv[0][r] *= alpha; accv[1][r] *= alpha;
            accv[2][r] *= alpha; accv[3][r] *= alpha;
            const int rowl = r + hs * 8;
            Ps[wid * 512 + rowl * 32 + lm]      = f2bf(p0);
            Ps[wid * 512 + rowl * 32 + 16 + lm] = f2bf(p1);
        }

        // P (16x32) as A fragment from per-wave LDS scratch
        Frag16 pf;
        {
            const uint16_t* p = &Ps[wid * 512 + lm * 32 + hs * 8];
            pf.u[0] = *(const uint4*)p;
            pf.u[1] = *(const uint4*)(p + 16);
        }
        // ctx += P * V  (V^T staged so B fragments are contiguous per lane)
#pragma unroll
        for (int ni = 0; ni < 4; ++ni) {
            Frag16 vb;
            const uint16_t* p = &Vts[(ni * 16 + lm) * 32 + hs * 8];
            vb.u[0] = *(const uint4*)p;
            vb.u[1] = *(const uint4*)(p + 16);
            accv[ni] = wmma_bf16(pf.v, vb.v, accv[ni]);
        }
    }

    // normalize and store ctx (bf16)
#pragma unroll
    for (int r = 0; r < 8; ++r) {
        const float inv = 1.0f / l_run[r];
        const int row = qrowg + r + hs * 8;
#pragma unroll
        for (int ni = 0; ni < 4; ++ni)
            ctx[(size_t)row * 1024 + h * 64 + ni * 16 + lm] = f2bf(accv[ni][r] * inv);
    }
}

// ---------------------------------------------------------------------------
// launch
// ---------------------------------------------------------------------------
extern "C" void kernel_launch(void* const* d_in, const int* in_sizes, int n_in,
                              void* d_out, int out_size, void* d_ws, size_t ws_size,
                              hipStream_t stream) {
    const float* x      = (const float*)d_in[0];   // [4096,1024]
    const float* wqkv   = (const float*)d_in[1];   // [3072,1024]
    const float* wdense = (const float*)d_in[2];   // [1024,1024]
    float* out = (float*)d_out;

    char* ws = (char*)d_ws;
    uint16_t* xb      = (uint16_t*)(ws);                        //  8 MB
    uint16_t* wqkvb   = (uint16_t*)(ws + 8388608);              //  6 MB
    uint16_t* wdenseb = (uint16_t*)(ws + 14680064);             //  2 MB
    uint16_t* qkvb    = (uint16_t*)(ws + 16777216);             // 24 MB [4096,3072]
    uint16_t* ctxb    = (uint16_t*)(ws + 41943040);             //  8 MB [4096,1024]

    cvt_f32_bf16<<<4096, 256, 0, stream>>>(x,      xb,      4096 * 1024);
    cvt_f32_bf16<<<3072, 256, 0, stream>>>(wqkv,   wqkvb,   3072 * 1024);
    cvt_f32_bf16<<<1024, 256, 0, stream>>>(wdense, wdenseb, 1024 * 1024);

    // QKV projection: [4096,1024] x [3072,1024]^T -> bf16 [4096,3072]
    gemm_bf16_nt<false><<<dim3(24, 32), 256, 0, stream>>>(
        xb, wqkvb, qkvb, 4096, 3072, 1024);

    // Attention: 2 batches * 16 heads * 16 query-blocks of 128
    attn_fwd<<<512, 256, 0, stream>>>(qkvb, ctxb);

    // Output dense: [4096,1024] x [1024,1024]^T -> f32 out
    gemm_bf16_nt<true><<<dim3(8, 32), 256, 0, stream>>>(
        ctxb, wdenseb, out, 4096, 1024, 1024);
}
